// RecurrentNeuroSymbolicCell_24902220382571
// MI455X (gfx1250) — matrix operand order repaired and checked by reading, not compiled, not run on hardware
//
#include <hip/hip_runtime.h>
#include <math.h>

// ---------------------------------------------------------------- types
typedef __attribute__((ext_vector_type(16))) __bf16 bf16x16;
typedef __attribute__((ext_vector_type(8)))  float  f32x8;

struct alignas(16) B128 { unsigned int w[4]; };
union FragU { bf16x16 v; B128 b[2]; };

static constexpr int BB = 32, LL = 512, DD = 256, HDx = 32, VV = 4096, KK = 128;
static constexpr int MM = BB * LL;   // 16384 token rows

#define WMMA_BF16(a, b, c) \
  __builtin_amdgcn_wmma_f32_16x16x32_bf16(false, (a), false, (b), (short)0, (c), false, false)

// -------------------------------------------------- fragment loaders (ISA 7.12.2)
// A 16x32 bf16, row-major source, lda elements.
__device__ inline bf16x16 load_frag_a(const __bf16* A, int lda, int m0, int k0, int lane) {
  int row = m0 + (lane & 15);
  int kb  = (lane < 16) ? 0 : 8;
  const __bf16* p = A + (size_t)row * lda + k0 + kb;
  FragU u;
  u.b[0] = *reinterpret_cast<const B128*>(p);        // K = kb .. kb+7
  u.b[1] = *reinterpret_cast<const B128*>(p + 16);   // K = 16+kb .. 16+kb+7
  return u.v;
}

// B 32x16 bf16 where B[k][n] = W[n][k]; W row-major (N x Kd), ldw elements.
__device__ inline bf16x16 load_frag_w(const __bf16* W, int ldw, int n0, int k0, int lane) {
  int col = n0 + (lane & 15);
  int kb  = (lane < 16) ? 0 : 8;
  const __bf16* p = W + (size_t)col * ldw + k0 + kb;
  FragU u;
  u.b[0] = *reinterpret_cast<const B128*>(p);        // K = kb .. kb+7
  u.b[1] = *reinterpret_cast<const B128*>(p + 16);   // K = 16+kb .. 16+kb+7
  return u.v;
}

// ---------------------------------------------------------------- generic GEMM
// C[m,n] = epi( sum_k A[m,k]*W[n,k] + bias[n] ), A: M x Kd bf16, W: N x Kd bf16.
// Per wave: 16(M) x 64(N); block = 4 waves => 64 x 64 tile.
// epi: 0 = f32 store, 1 = bf16 store,
//      2 = sigmoid(v + bias + rowbias[(row/rowdiv)*N+n]) f32,
//      3 = bf16 store transposed per-head for PV: dst[((b*8+h)*32+hd)*512 + l]
//          (requires N==256, row tiles not crossing a batch of 512 rows).
__global__ __launch_bounds__(128) void gemm_bt_kernel(
    const __bf16* __restrict__ A, const __bf16* __restrict__ W,
    const float* __restrict__ bias, const float* __restrict__ rowbias,
    float* __restrict__ Cf, __bf16* __restrict__ Cb,
    int N, int Kd, int epi, int rowdiv) {
  const int lane = threadIdx.x & 31;
  const int wave = threadIdx.x >> 5;
  const int m0 = blockIdx.x * 64 + wave * 16;
  const int n0 = blockIdx.y * 64;
  f32x8 acc[4] = {};
  for (int k0 = 0; k0 < Kd; k0 += 32) {
    bf16x16 a = load_frag_a(A, Kd, m0, k0, lane);
#pragma unroll
    for (int t = 0; t < 4; ++t) {
      bf16x16 b = load_frag_w(W, Kd, n0 + t * 16, k0, lane);
      acc[t] = WMMA_BF16(a, b, acc[t]);
    }
  }
  const int colb = lane & 15;
  const int rb   = (lane < 16) ? 0 : 8;
#pragma unroll
  for (int t = 0; t < 4; ++t) {
    int col = n0 + t * 16 + colb;
    float bv = bias ? bias[col] : 0.f;
    if (epi == 3) {
      union { B128 raw; __bf16 h[8]; } pk;
#pragma unroll
      for (int v = 0; v < 8; ++v) pk.h[v] = (__bf16)(acc[t][v] + bv);
      // lane's 8 rows are consecutive l values -> one 16B store
      size_t o = (size_t)(((m0 >> 9) * 8 + (col >> 5)) * 32 + (col & 31)) * LL
               + (m0 & 511) + rb;
      *reinterpret_cast<B128*>(Cb + o) = pk.raw;
    } else {
#pragma unroll
      for (int v = 0; v < 8; ++v) {
        int row = m0 + rb + v;
        float val = acc[t][v] + bv;
        size_t o = (size_t)row * N + col;
        if (epi == 0)      Cf[o] = val;
        else if (epi == 1) Cb[o] = (__bf16)val;
        else {
          val += rowbias[(row / rowdiv) * N + col];
          Cf[o] = 1.f / (1.f + __expf(-val));
        }
      }
    }
  }
}

// ------------------------------------------------------- complex out-projection
// Cr = Ar*Wr^T - Ai*Wi^T + bm ; Ci = Ai*Wr^T + Ar*Wi^T + bp
__global__ __launch_bounds__(128) void cgemm_kernel(
    const __bf16* __restrict__ Ar, const __bf16* __restrict__ Ai,
    const __bf16* __restrict__ Wr, const __bf16* __restrict__ Wi,
    const float* __restrict__ bm, const float* __restrict__ bp,
    float* __restrict__ Cr, float* __restrict__ Ci, int N, int Kd) {
  const int lane = threadIdx.x & 31;
  const int wave = threadIdx.x >> 5;
  const int m0 = blockIdx.x * 64 + wave * 16;
  const int n0 = blockIdx.y * 16;
  f32x8 arr = {}, aii = {}, air = {}, ari = {};
  for (int k0 = 0; k0 < Kd; k0 += 32) {
    bf16x16 ar = load_frag_a(Ar, Kd, m0, k0, lane);
    bf16x16 ai = load_frag_a(Ai, Kd, m0, k0, lane);
    bf16x16 wr = load_frag_w(Wr, Kd, n0, k0, lane);
    bf16x16 wi = load_frag_w(Wi, Kd, n0, k0, lane);
    arr = WMMA_BF16(ar, wr, arr);
    aii = WMMA_BF16(ai, wi, aii);
    air = WMMA_BF16(ai, wr, air);
    ari = WMMA_BF16(ar, wi, ari);
  }
  const int col = n0 + (lane & 15);
  const int rb  = (lane < 16) ? 0 : 8;
#pragma unroll
  for (int v = 0; v < 8; ++v) {
    int row = m0 + rb + v;
    size_t o = (size_t)row * N + col;
    Cr[o] = arr[v] - aii[v] + bm[col];
    Ci[o] = air[v] + ari[v] + bp[col];
  }
}

// ------------------------------------------------------------- fused attention
// One wave per (b, h, 16-query-row tile). scores(16x512) in LDS, softmax,
// probs->bf16 in LDS, PV via WMMA against per-head transposed V.
// scores = (qr.kr^T + qi.ki^T)/sqrt(32).
__global__ __launch_bounds__(32) void attention_kernel(
    const __bf16* __restrict__ qr, const __bf16* __restrict__ qi,
    const __bf16* __restrict__ kr, const __bf16* __restrict__ ki,
    const __bf16* __restrict__ vtr, const __bf16* __restrict__ vti,
    __bf16* __restrict__ aro, __bf16* __restrict__ aio) {
  __shared__ __align__(16) float  sc[16 * LL];
  __shared__ __align__(16) __bf16 pb[16 * LL];
  const int lane = threadIdx.x;
  const int bid  = blockIdx.x;
  const int mt = bid & 31;
  const int h  = (bid >> 5) & 7;
  const int b  = bid >> 8;
  const int rowg = b * LL + mt * 16;
  const float scale = 0.17677669529663687f;  // 1/sqrt(32)
  const int colb = lane & 15;
  const int rb   = (lane < 16) ? 0 : 8;

  bf16x16 aR = load_frag_a(qr, DD, rowg, h * HDx, lane);
  bf16x16 aI = load_frag_a(qi, DD, rowg, h * HDx, lane);

  for (int st = 0; st < LL / 16; ++st) {
    bf16x16 bR = load_frag_w(kr, DD, b * LL + st * 16, h * HDx, lane);
    bf16x16 bI = load_frag_w(ki, DD, b * LL + st * 16, h * HDx, lane);
    f32x8 acc = {};
    acc = WMMA_BF16(aR, bR, acc);
    acc = WMMA_BF16(aI, bI, acc);
#pragma unroll
    for (int v = 0; v < 8; ++v) sc[(rb + v) * LL + st * 16 + colb] = acc[v] * scale;
  }
  __syncthreads();

  {  // softmax: lane pair (r, r+16) handles row r, 256 cols each
    const int r = lane & 15;
    const int half = lane >> 4;
    float* rowp = sc + r * LL + half * 256;
    float mx = -1e30f;
    for (int c = 0; c < 256; ++c) mx = fmaxf(mx, rowp[c]);
    mx = fmaxf(mx, __shfl_xor(mx, 16));
    float sum = 0.f;
    for (int c = 0; c < 256; ++c) { float e = __expf(rowp[c] - mx); rowp[c] = e; sum += e; }
    sum += __shfl_xor(sum, 16);
    float inv = 1.f / sum;
    __bf16* prow = pb + r * LL + half * 256;
    for (int c = 0; c < 256; ++c) prow[c] = (__bf16)(rowp[c] * inv);
  }
  __syncthreads();

  // attn = probs @ v  via vT[b,h] (32 x 512 row-major): B[k][n] = vT[n][k]
  const __bf16* vtrh = vtr + (size_t)(b * 8 + h) * HDx * LL;
  const __bf16* vtih = vti + (size_t)(b * 8 + h) * HDx * LL;
  f32x8 ar0 = {}, ar1 = {}, ai0 = {}, ai1 = {};
  for (int ks = 0; ks < LL / 32; ++ks) {
    bf16x16 ap = load_frag_a(pb, LL, 0, ks * 32, lane);
    bf16x16 b0 = load_frag_w(vtrh, LL, 0,  ks * 32, lane);
    bf16x16 b1 = load_frag_w(vtrh, LL, 16, ks * 32, lane);
    bf16x16 c0 = load_frag_w(vtih, LL, 0,  ks * 32, lane);
    bf16x16 c1 = load_frag_w(vtih, LL, 16, ks * 32, lane);
    ar0 = WMMA_BF16(ap, b0, ar0);
    ar1 = WMMA_BF16(ap, b1, ar1);
    ai0 = WMMA_BF16(ap, c0, ai0);
    ai1 = WMMA_BF16(ap, c1, ai1);
  }
#pragma unroll
  for (int v = 0; v < 8; ++v) {
    size_t o = (size_t)(rowg + rb + v) * DD + h * HDx + colb;
    aro[o]      = (__bf16)ar0[v];
    aro[o + 16] = (__bf16)ar1[v];
    aio[o]      = (__bf16)ai0[v];
    aio[o + 16] = (__bf16)ai1[v];
  }
}

// ---------------------------------------------------------------- small kernels
__global__ void cvt_bf16_kernel(const float* __restrict__ s, __bf16* __restrict__ d, int n) {
  int i = blockIdx.x * 256 + threadIdx.x;
  if (i < n) d[i] = (__bf16)s[i];
}

__global__ void embed_kernel(const int* __restrict__ tok, const float* __restrict__ emb,
                             __bf16* __restrict__ x) {
  int i = blockIdx.x * 256 + threadIdx.x;
  int m = i >> 8, d = i & 255;
  x[i] = (__bf16)emb[(size_t)tok[m] * DD + d];
}

// Ge[n][d] = gate_w[n][2d]  (real slots of interleaved view_as_real block)
__global__ void make_ge_kernel(const float* __restrict__ gw, __bf16* __restrict__ ge) {
  int i = blockIdx.x * 256 + threadIdx.x;
  int n = i >> 8, d = i & 255;
  ge[i] = (__bf16)gw[n * 768 + 2 * d];
}

__global__ void prep_bias_kernel(const float* __restrict__ br, const float* __restrict__ bi,
                                 float* __restrict__ bm, float* __restrict__ bp) {
  int i = blockIdx.x * 256 + threadIdx.x;  // 4*256
  bm[i] = br[i] - bi[i];
  bp[i] = br[i] + bi[i];
}

__global__ void vq_norm_kernel(const float* __restrict__ vq, float* __restrict__ c) {
  int k = threadIdx.x;  // 128
  float s = 0.f;
  for (int d = 0; d < DD; ++d) { float t = vq[k * DD + d]; s += t * t; }
  c[k] = s;
}

// cg[b][n] = sum_d ctx[b][d] * gate_w[n][512 + d]
__global__ void cgate_kernel(const float* __restrict__ ctx, const float* __restrict__ gw,
                             float* __restrict__ cg) {
  int b = blockIdx.x, n = threadIdx.x;
  float s = 0.f;
  for (int d = 0; d < DD; ++d) s += ctx[b * DD + d] * gw[n * 768 + 512 + d];
  cg[b * DD + n] = s;
}

__global__ void argmin_kernel(const float* __restrict__ G, const float* __restrict__ c,
                              int* __restrict__ idx) {
  int m = blockIdx.x * 256 + threadIdx.x;
  float best = 3.4e38f; int bi = 0;
  for (int k = 0; k < KK; ++k) {
    float dv = c[k] - 2.f * G[(size_t)m * KK + k];
    if (dv < best) { best = dv; bi = k; }
  }
  idx[m] = bi;
}

__global__ void gather_zq_kernel(const int* __restrict__ idx, const __bf16* __restrict__ vqb,
                                 __bf16* __restrict__ zq) {
  int i = blockIdx.x * 256 + threadIdx.x;
  int m = i >> 8, d = i & 255;
  zq[i] = vqb[idx[m] * DD + d];
}

__global__ void stack_kernel(const float* __restrict__ zg, float* __restrict__ st,
                             float* __restrict__ tail) {
  int b = blockIdx.x, d = threadIdx.x;
  float s = 0.f;
  const float* p = zg + (size_t)b * LL * DD + d;
  for (int l = 0; l < LL; ++l) s += p[(size_t)l * DD];
  st[b * DD + d] = s;
  tail[b * DD + d] = s;
}

__global__ void ln_modrelu_kernel(const float* __restrict__ zg, const float* __restrict__ st,
                                  const float* __restrict__ g, const float* __restrict__ bR,
                                  const float* __restrict__ bI, const float* __restrict__ mrb,
                                  __bf16* __restrict__ outr) {
  __shared__ float sA[256], sB[256];
  int m = blockIdx.x, d = threadIdx.x;
  float x = zg[(size_t)m * DD + d] + st[(m >> 9) * DD + d];
  sA[d] = x; sB[d] = x * x;
  __syncthreads();
  for (int s = 128; s > 0; s >>= 1) {
    if (d < s) { sA[d] += sA[d + s]; sB[d] += sB[d + s]; }
    __syncthreads();
  }
  float mu  = sA[0] * (1.f / 256.f);
  float var = sB[0] * (1.f / 256.f) - mu * mu;
  float rstd = rsqrtf(var + 1e-5f);
  float nr = (x - mu) * rstd * g[d] + bR[d];
  float ni = bI[d];                       // layernorm(zeros) == bias
  float mag = sqrtf(nr * nr + ni * ni);
  float sc2 = fmaxf(mag + mrb[d], 0.f) / (mag + 1e-6f);
  outr[(size_t)m * DD + d] = (__bf16)(nr * sc2);
}

__global__ void zero1_kernel(float* p) { *p = 0.f; }

__global__ void loss_kernel(const float* __restrict__ zar, const float* __restrict__ zai,
                            const int* __restrict__ idx, const float* __restrict__ vq,
                            float* __restrict__ loss) {
  __shared__ float red[256];
  int i = blockIdx.x * 256 + threadIdx.x;  // over B*L*D
  int m = i >> 8, d = i & 255;
  float dr = vq[(size_t)idx[m] * DD + d] - zar[i];
  float di = zai[i];
  red[threadIdx.x] = dr * dr + di * di;
  __syncthreads();
  for (int s = 128; s > 0; s >>= 1) {
    if (threadIdx.x < s) red[threadIdx.x] += red[threadIdx.x + s];
    __syncthreads();
  }
  if (threadIdx.x == 0) atomicAdd(loss, red[0] * (2.f / (float)(MM * DD)));
}

// ---------------------------------------------------------------- launcher
extern "C" void kernel_launch(void* const* d_in, const int* in_sizes, int n_in,
                              void* d_out, int out_size, void* d_ws, size_t ws_size,
                              hipStream_t stream) {
  (void)in_sizes; (void)n_in; (void)out_size; (void)ws_size;
  const int*   tokens  = (const int*)d_in[0];
  const float* context = (const float*)d_in[1];
  const float* emb     = (const float*)d_in[2];
  const float* attn_wr = (const float*)d_in[3];
  const float* attn_wi = (const float*)d_in[4];
  const float* attn_br = (const float*)d_in[5];
  const float* attn_bi = (const float*)d_in[6];
  const float* vq_emb  = (const float*)d_in[7];
  const float* gate_w  = (const float*)d_in[8];
  const float* gate_b  = (const float*)d_in[9];
  const float* ln_r_g  = (const float*)d_in[10];
  const float* ln_r_b  = (const float*)d_in[11];
  /* ln_i_g (d_in[12]) multiplies normalized zeros -> unused */
  const float* ln_i_b  = (const float*)d_in[13];
  const float* modrelu = (const float*)d_in[14];
  const float* head_w  = (const float*)d_in[15];
  const float* head_b  = (const float*)d_in[16];
  float* out = (float*)d_out;

  char* w = (char*)d_ws;
  // persistent small buffers
  __bf16* WRB = (__bf16*)(w + 0);          // 4*D*D bf16        (524288 B)
  __bf16* WIB = (__bf16*)(w + 524288);     // 4*D*D bf16
  __bf16* HWB = (__bf16*)(w + 1048576);    // V*D bf16          (2 MB)
  __bf16* VQB = (__bf16*)(w + 3145728);    // K*D bf16
  __bf16* GEB = (__bf16*)(w + 3211264);    // D*D bf16
  float*  BM  = (float*)(w + 3342336);     // 4*D
  float*  BP  = (float*)(w + 3346432);     // 4*D
  float*  VQC = (float*)(w + 3350528);     // K
  float*  CG  = (float*)(w + 3351552);     // B*D
  int*    IDX = (int*)(w + 3384320);       // M
  float*  ST  = (float*)(w + 3449856);     // B*D
  float*  ZAR = (float*)(w + 3482624);     // M*D f32 (16 MB)
  float*  ZAI = (float*)(w + 20259840);    // M*D f32
  __bf16* ARB = (__bf16*)(w + 37037056);   // M*D bf16 (8 MB)
  __bf16* AIB = (__bf16*)(w + 45425664);   // M*D bf16
  // reusable 8 MB regions R0..R6
  char* R0 = w + 53814272;   // x_bf       -> zr_bf
  char* R1 = w + 62202880;   // qr         -> zq_bf
  char* R2 = w + 70591488;   // qi         -> G (M*K f32)
  char* R3 = w + 78980096;   // kr         -> z_gated (f32, spans R3+R4)
  char* R4 = w + 87368704;   // ki
  char* R5 = w + 95757312;   // vT_r       -> outr_bf
  char* R6 = w + 104145920;  // vT_i
  __bf16* XBF = (__bf16*)R0;
  __bf16* QR = (__bf16*)R1, *QI = (__bf16*)R2, *KR = (__bf16*)R3;
  __bf16* KI = (__bf16*)R4, *VTR = (__bf16*)R5, *VTI = (__bf16*)R6;

  const int DxD = DD * DD;

  // ---- prep: weight conversions, derived biases, codebook norms, ctx gate
  cvt_bf16_kernel<<<dim3(4 * DxD / 256), dim3(256), 0, stream>>>(attn_wr, WRB, 4 * DxD);
  cvt_bf16_kernel<<<dim3(4 * DxD / 256), dim3(256), 0, stream>>>(attn_wi, WIB, 4 * DxD);
  cvt_bf16_kernel<<<dim3(VV * DD / 256), dim3(256), 0, stream>>>(head_w, HWB, VV * DD);
  cvt_bf16_kernel<<<dim3(KK * DD / 256), dim3(256), 0, stream>>>(vq_emb, VQB, KK * DD);
  make_ge_kernel<<<dim3(DxD / 256), dim3(256), 0, stream>>>(gate_w, GEB);
  prep_bias_kernel<<<dim3(4), dim3(256), 0, stream>>>(attn_br, attn_bi, BM, BP);
  vq_norm_kernel<<<dim3(1), dim3(128), 0, stream>>>(vq_emb, VQC);
  cgate_kernel<<<dim3(BB), dim3(256), 0, stream>>>(context, gate_w, CG);

  // ---- embedding gather -> bf16
  embed_kernel<<<dim3(MM * DD / 256), dim3(256), 0, stream>>>(tokens, emb, XBF);

  // ---- QKV projections (input imag == 0): re = x@Wr^T + (br-bi), im = x@Wi^T + (br+bi)
  // V outputs are stored transposed per head (epi=3) to feed the PV WMMA stage.
  dim3 gD(MM / 64, DD / 64), blk(128);
  gemm_bt_kernel<<<gD, blk, 0, stream>>>(XBF, WRB,           BM,       nullptr, nullptr, QR,  DD, DD, 1, 1);
  gemm_bt_kernel<<<gD, blk, 0, stream>>>(XBF, WIB,           BP,       nullptr, nullptr, QI,  DD, DD, 1, 1);
  gemm_bt_kernel<<<gD, blk, 0, stream>>>(XBF, WRB + DxD,     BM + 256, nullptr, nullptr, KR,  DD, DD, 1, 1);
  gemm_bt_kernel<<<gD, blk, 0, stream>>>(XBF, WIB + DxD,     BP + 256, nullptr, nullptr, KI,  DD, DD, 1, 1);
  gemm_bt_kernel<<<gD, blk, 0, stream>>>(XBF, WRB + 2 * DxD, BM + 512, nullptr, nullptr, VTR, DD, DD, 3, 1);
  gemm_bt_kernel<<<gD, blk, 0, stream>>>(XBF, WIB + 2 * DxD, BP + 512, nullptr, nullptr, VTI, DD, DD, 3, 1);

  // ---- fused attention
  attention_kernel<<<dim3(BB * 8 * (LL / 16)), dim3(32), 0, stream>>>(QR, QI, KR, KI, VTR, VTI, ARB, AIB);

  // ---- complex out-projection
  cgemm_kernel<<<dim3(MM / 64, DD / 16), blk, 0, stream>>>(
      ARB, AIB, WRB + 3 * DxD, WIB + 3 * DxD, BM + 768, BP + 768, ZAR, ZAI, DD, DD);

  // ---- VQ: zr->bf16, G = zr@vq^T, argmin(-2G + ||e||^2), gather codebook rows
  __bf16* ZRB = (__bf16*)R0;
  float*  G   = (float*)R2;
  __bf16* ZQB = (__bf16*)R1;
  cvt_bf16_kernel<<<dim3(MM * DD / 256), dim3(256), 0, stream>>>(ZAR, ZRB, MM * DD);
  gemm_bt_kernel<<<dim3(MM / 64, KK / 64), blk, 0, stream>>>(ZRB, VQB, nullptr, nullptr, G, nullptr, KK, DD, 0, 1);
  argmin_kernel<<<dim3(MM / 256), dim3(256), 0, stream>>>(G, VQC, IDX);
  gather_zq_kernel<<<dim3(MM * DD / 256), dim3(256), 0, stream>>>(IDX, VQB, ZQB);

  // ---- gate: sigmoid( zq@Ge^T + ctx@Gc^T + gate_b )
  float* ZG = (float*)R3;  // 16 MB spans R3+R4
  gemm_bt_kernel<<<gD, blk, 0, stream>>>(ZQB, GEB, gate_b, CG, ZG, nullptr, DD, DD, 2, LL);

  // ---- stack top (sum over L) -> ST and output tail
  float* tail = out + (size_t)MM * VV + 1;
  stack_kernel<<<dim3(BB), dim3(256), 0, stream>>>(ZG, ST, tail);

  // ---- layernorm + modrelu -> bf16 head input
  __bf16* OUTB = (__bf16*)R5;
  ln_modrelu_kernel<<<dim3(MM), dim3(256), 0, stream>>>(ZG, ST, ln_r_g, ln_r_b, ln_i_b, modrelu, OUTB);

  // ---- vocab head -> logits
  gemm_bt_kernel<<<dim3(MM / 64, VV / 64), blk, 0, stream>>>(OUTB, HWB, head_b, nullptr, out, nullptr, VV, DD, 0, 1);

  // ---- vq loss scalar
  float* lossp = out + (size_t)MM * VV;
  zero1_kernel<<<dim3(1), dim3(1), 0, stream>>>(lossp);
  loss_kernel<<<dim3(MM * DD / 256), dim3(256), 0, stream>>>(ZAR, ZAI, IDX, vq_emb, lossp);
}
